// MixedSparseTraditionalMLP_75153337745537
// MI455X (gfx1250) — compile-verified, matrix operand order checked by reading, not run on hardware
//
#include <hip/hip_runtime.h>
#include <hip/hip_bf16.h>

typedef __attribute__((ext_vector_type(8)))  float  v8f;
typedef __attribute__((ext_vector_type(16))) __bf16 v16bf;
typedef __attribute__((ext_vector_type(4)))  int    v4i;
typedef __attribute__((address_space(1))) v4i* as1_v4i_ptr;
typedef __attribute__((address_space(3))) v4i* as3_v4i_ptr;

union Frag { v16bf v; uint4 q[2]; };

#if __has_builtin(__builtin_amdgcn_global_load_async_to_lds_b128)
#define USE_ASYNC 1
#else
#define USE_ASYNC 0
#endif

__device__ __forceinline__ void wait_async0() {
#if __has_builtin(__builtin_amdgcn_s_wait_asynccnt)
    __builtin_amdgcn_s_wait_asynccnt(0);
#else
    asm volatile("s_wait_asynccnt 0" ::: "memory");
#endif
}

__device__ __forceinline__ void async_cp16(const void* g, void* l) {
#if USE_ASYNC
    __builtin_amdgcn_global_load_async_to_lds_b128(
        (as1_v4i_ptr)(v4i*)g, (as3_v4i_ptr)(v4i*)l, 0, 0);
#else
    *reinterpret_cast<uint4*>(l) = *reinterpret_cast<const uint4*>(g);
#endif
}

__device__ __forceinline__ unsigned short f2bf(float f) {
    unsigned u = __float_as_uint(f);
    unsigned r = u + 0x7FFFu + ((u >> 16) & 1u);   // round-to-nearest-even
    return (unsigned short)(r >> 16);
}

// ---------------- prep kernels ----------------

__global__ void zero_i32(int* __restrict__ p, int n) {
    int i = blockIdx.x * blockDim.x + threadIdx.x;
    if (i < n) p[i] = 0;
}

__global__ void cvt_bf16(const float* __restrict__ in, unsigned short* __restrict__ out, int n) {
    int i = blockIdx.x * blockDim.x + threadIdx.x;
    int j = i * 2;
    if (j < n) {
        float2 f = *reinterpret_cast<const float2*>(in + j);
        unsigned pack = (unsigned)f2bf(f.x) | ((unsigned)f2bf(f.y) << 16);
        *reinterpret_cast<unsigned*>(out + j) = pack;
    }
}

// out[n][k] = W[k][n] + sum_r La[k][r]*Lb[r][n]   (bf16, transposed to N-major)
__global__ __launch_bounds__(1024)
void weff_T(const float* __restrict__ W, const float* __restrict__ La,
            const float* __restrict__ Lb, unsigned short* __restrict__ out,
            int K, int N)
{
    __shared__ unsigned short tile[32][33];
    __shared__ float As[32][16];   // [k][r]
    __shared__ float Bs[16][32];   // [r][n]
    const int tx = threadIdx.x, ty = threadIdx.y;
    const int k0 = blockIdx.y * 32, n0 = blockIdx.x * 32;
    if (ty < 16) {
        As[tx][ty] = La[(size_t)(k0 + tx) * 16 + ty];
        Bs[ty][tx] = Lb[(size_t)ty * N + n0 + tx];
    }
    __syncthreads();
    float val = W[(size_t)(k0 + ty) * N + n0 + tx];
#pragma unroll
    for (int r = 0; r < 16; ++r) val += As[ty][r] * Bs[r][tx];
    tile[ty][tx] = f2bf(val);
    __syncthreads();
    out[(size_t)(n0 + ty) * K + k0 + tx] = tile[tx][ty];
}

// ---------------- GEMM (bf16 WMMA, f32 accum, async global->LDS pipeline) ----
// C[M,N] = A[M,K] * Bt[N,K]^T ; mode 1: relu + f32 & bf16 stores + zero counts
#define BM 128
#define BN 64
#define BK 64
#define LSTR 72   // padded LDS row stride (elements)

__global__ __launch_bounds__(256)
void gemm_bf16(const unsigned short* __restrict__ A,
               const unsigned short* __restrict__ Bt,
               float* __restrict__ Cf,
               unsigned short* __restrict__ Cb,
               int* __restrict__ zc,
               int M, int N, int K, int mode)
{
    __shared__ unsigned short ldsA[2][BM * LSTR];
    __shared__ unsigned short ldsB[2][BN * LSTR];
    __shared__ int lzc[BN];

    const int tid   = threadIdx.x;
    const int lane  = tid & 31;
    const int wid   = tid >> 5;
    const int lmod  = lane & 15;
    const int lhalf = lane >> 4;
    const int waveM = (wid & 3) * 32;
    const int waveN = (wid >> 2) * 32;
    const int bm = blockIdx.y * BM;
    const int bn = blockIdx.x * BN;
    const bool doAux = (mode == 1);

    if (tid < BN) lzc[tid] = 0;

    const v8f vzero = {0.f,0.f,0.f,0.f,0.f,0.f,0.f,0.f};
    v8f acc[2][2];
#pragma unroll
    for (int i = 0; i < 2; ++i)
#pragma unroll
        for (int j = 0; j < 2; ++j) acc[i][j] = vzero;

    // stage one BK-chunk (A: 128x64, B: 64x64 bf16) into LDS buffer (t&1)
    auto issue = [&](int t) {
        const int kk = t * BK;
        const int bsel = t & 1;
#pragma unroll
        for (int i = 0; i < 4; ++i) {               // A: 1024 x b128 chunks / 256 thr
            int c = tid + i * 256;
            int row = c >> 3, kc = c & 7;
            async_cp16(A + (size_t)(bm + row) * K + kk + kc * 8,
                       &ldsA[bsel][row * LSTR + kc * 8]);
        }
#pragma unroll
        for (int i = 0; i < 2; ++i) {               // B: 512 x b128 chunks / 256 thr
            int c = tid + i * 256;
            int row = c >> 3, kc = c & 7;
            async_cp16(Bt + (size_t)(bn + row) * K + kk + kc * 8,
                       &ldsB[bsel][row * LSTR + kc * 8]);
        }
    };

    const int T = K / BK;
    issue(0);
    for (int t = 0; t < T; ++t) {
#if USE_ASYNC
        wait_async0();
#endif
        __syncthreads();
        if (t + 1 < T) issue(t + 1);
        const int bsel = t & 1;
#pragma unroll
        for (int ks = 0; ks < 2; ++ks) {
            const int kOff = ks * 32 + lhalf * 8;   // ISA 16-bit A/B fragment layout
            Frag fa[2], fb[2];
#pragma unroll
            for (int mi = 0; mi < 2; ++mi) {
                int row = waveM + mi * 16 + lmod;
                fa[mi].q[0] = *reinterpret_cast<const uint4*>(&ldsA[bsel][row * LSTR + kOff]);
                fa[mi].q[1] = *reinterpret_cast<const uint4*>(&ldsA[bsel][row * LSTR + kOff + 16]);
            }
#pragma unroll
            for (int ni = 0; ni < 2; ++ni) {
                int col = waveN + ni * 16 + lmod;
                fb[ni].q[0] = *reinterpret_cast<const uint4*>(&ldsB[bsel][col * LSTR + kOff]);
                fb[ni].q[1] = *reinterpret_cast<const uint4*>(&ldsB[bsel][col * LSTR + kOff + 16]);
            }
#pragma unroll
            for (int mi = 0; mi < 2; ++mi)
#pragma unroll
                for (int ni = 0; ni < 2; ++ni)
                    acc[mi][ni] = __builtin_amdgcn_wmma_f32_16x16x32_bf16(
                        false, fa[mi].v, false, fb[ni].v,
                        (short)0, acc[mi][ni], false, false);
        }
    }

    // epilogue: C layout -> VGPR v: M = base + 8*lhalf + v, N = base + lmod
#pragma unroll
    for (int mi = 0; mi < 2; ++mi) {
#pragma unroll
        for (int ni = 0; ni < 2; ++ni) {
            const int n = bn + waveN + ni * 16 + lmod;
            const int mbase = bm + waveM + mi * 16 + lhalf * 8;
            int cnt = 0;
#pragma unroll
            for (int v = 0; v < 8; ++v) {
                float val = acc[mi][ni][v];
                size_t off = (size_t)(mbase + v) * N + n;
                if (doAux) {
                    val = val > 0.f ? val : 0.f;
                    cnt += (val == 0.f) ? 1 : 0;
                    Cf[off] = val;
                    Cb[off] = f2bf(val);
                } else {
                    Cf[off] = val;
                }
            }
            if (doAux) atomicAdd(&lzc[waveN + ni * 16 + lmod], cnt);
        }
    }
    if (doAux) {
        __syncthreads();
        if (tid < BN) {
            int b = bm / 2048;                  // batch of this 128-row block
            atomicAdd(&zc[(size_t)b * 8192 + bn + tid], lzc[tid]);
        }
    }
}

// ---------------- stable top-k via LDS bitonic sort ----------------
__global__ __launch_bounds__(1024)
void topk_kernel(const int* __restrict__ zc, int* __restrict__ outIdx)
{
    __shared__ unsigned keys[8192];
    const int b = blockIdx.x;
    const int tid = threadIdx.x;
    for (int i = tid; i < 8192; i += 1024)
        keys[i] = (((unsigned)zc[b * 8192 + i]) << 13) | (unsigned)i;
    __syncthreads();
    for (int k = 2; k <= 8192; k <<= 1) {
        for (int j = k >> 1; j > 0; j >>= 1) {
            for (int i = tid; i < 8192; i += 1024) {
                int ixj = i ^ j;
                if (ixj > i) {
                    unsigned a = keys[i], c = keys[ixj];
                    bool up = ((i & k) == 0);
                    if ((a > c) == up) { keys[i] = c; keys[ixj] = a; }
                }
            }
            __syncthreads();
        }
    }
    for (int r = tid; r < 4096; r += 1024)
        outIdx[b * 4096 + r] = (int)(keys[r] & 8191u);
}

// ---------------- column gather ----------------
__global__ void gather_kernel(const float* __restrict__ x2f,
                              const int* __restrict__ idx,
                              float* __restrict__ out)
{
    const int j = blockIdx.x * blockDim.x + threadIdx.x;  // 0..4095
    const int s = blockIdx.y;
    const int b = blockIdx.z;
    const int c = idx[b * 4096 + j];
    out[((size_t)b * 2048 + s) * 4096 + j] = x2f[((size_t)b * 2048 + s) * 8192 + c];
}

// ---------------- launch ----------------
extern "C" void kernel_launch(void* const* d_in, const int* in_sizes, int n_in,
                              void* d_out, int out_size, void* d_ws, size_t ws_size,
                              hipStream_t stream) {
    (void)in_sizes; (void)n_in; (void)out_size; (void)ws_size;
    const float* x    = (const float*)d_in[0];
    const float* wup  = (const float*)d_in[1];
    const float* wupA = (const float*)d_in[2];
    const float* wupB = (const float*)d_in[3];
    const float* wdn  = (const float*)d_in[4];
    const float* wdnA = (const float*)d_in[5];
    const float* wdnB = (const float*)d_in[6];

    const int B = 4, S = 2048, H = 2048, I = 8192;
    const int M = B * S;

    char* ws = (char*)d_ws;
    size_t off = 0;
    auto alloc = [&](size_t bytes) { void* p = ws + off; off += (bytes + 255) & ~(size_t)255; return p; };
    unsigned short* xb   = (unsigned short*)alloc((size_t)M * H * 2);
    unsigned short* wupT = (unsigned short*)alloc((size_t)I * H * 2);
    unsigned short* wdnT = (unsigned short*)alloc((size_t)H * I * 2);
    float*          x2f  = (float*)         alloc((size_t)M * I * 4);
    unsigned short* x2b  = (unsigned short*)alloc((size_t)M * I * 2);
    int*            zc   = (int*)           alloc((size_t)B * I * 4);

    float* y2    = (float*)d_out;
    float* x2top = y2 + (size_t)M * H;
    int*   idxO  = (int*)(x2top + (size_t)B * S * 4096);

    zero_i32<<<(B * I + 255) / 256, 256, 0, stream>>>(zc, B * I);
    cvt_bf16<<<(M * H / 2 + 255) / 256, 256, 0, stream>>>(x, xb, M * H);
    weff_T<<<dim3(I / 32, H / 32), dim3(32, 32), 0, stream>>>(wup, wupA, wupB, wupT, H, I);
    weff_T<<<dim3(H / 32, I / 32), dim3(32, 32), 0, stream>>>(wdn, wdnA, wdnB, wdnT, I, H);
    gemm_bf16<<<dim3(I / BN, M / BM), 256, 0, stream>>>(xb, wupT, x2f, x2b, zc, M, I, H, 1);
    topk_kernel<<<B, 1024, 0, stream>>>(zc, idxO);
    gemm_bf16<<<dim3(H / BN, M / BM), 256, 0, stream>>>(x2b, wdnT, y2, nullptr, nullptr, M, H, I, 0);
    gather_kernel<<<dim3(4096 / 256, S, B), 256, 0, stream>>>(x2f, idxO, x2top);
}
